// TemporalGCN_13331578487151
// MI455X (gfx1250) — compile-verified, matrix operand order checked by reading, not run on hardware
//
#include <hip/hip_runtime.h>

#define TT      50
#define NNODES  512
#define HDIM    128
#define NHEADS  4
#define HD      32
#define DINP    3
#define NPG     64

typedef __attribute__((ext_vector_type(16))) _Float16     v16h;
typedef __attribute__((ext_vector_type(8)))  float        v8f;
typedef __attribute__((ext_vector_type(4)))  unsigned int u32x4;
typedef __attribute__((ext_vector_type(8)))  int          i32x8;
typedef __attribute__((ext_vector_type(4)))  int          i32x4;

__device__ __forceinline__ int imin(int a, int b) { return a < b ? a : b; }

// CDNA5 16-bit A/B fragment layout (16x16x32): lane r = row (A:M / B:N),
// half = lane>>4; element e -> K = e + 8*half + 8*(e>=8). A fragment row is
// therefore two contiguous 8-half (16B) runs at half-offsets 8*half and
// 16+8*half -> two b128 loads.
union FragU { v16h v; u32x4 q[2]; };

__device__ __forceinline__ v16h frag_rowmajor_vec(const _Float16* base, int rstride,
                                                  int lane, int rows_valid) {
  int r = lane & 15, half = lane >> 4;
  FragU u;
  u32x4 z = {0u, 0u, 0u, 0u};
  u.q[0] = z; u.q[1] = z;
  if (r < rows_valid) {
    const _Float16* p = base + (size_t)r * rstride + 8 * half;
    u.q[0] = *(const u32x4*)(p);
    u.q[1] = *(const u32x4*)(p + 16);
  }
  return u.v;
}

__device__ __forceinline__ v8f wmma16(v16h a, v16h b, v8f c) {
  return __builtin_amdgcn_wmma_f32_16x16x32_f16(false, a, false, b,
                                                (short)0, c, false, false);
}

__device__ __forceinline__ v8f acc_zero() {
  v8f c;
#pragma unroll
  for (int i = 0; i < 8; ++i) c[i] = 0.0f;
  return c;
}

// ---------------- small prep kernels ----------------

__global__ void k_mask(const int* __restrict__ em, float* __restrict__ mf) {
  int idx = blockIdx.x * blockDim.x + threadIdx.x;
  if (idx >= TT * NNODES) return;
  int t = idx / NNODES, n = idx % NNODES;
  int b = n / NPG, g = n % NPG;
  mf[idx] = (em[(b * TT + t) * NPG + g] != 0) ? 1.0f : 0.0f;
}

__global__ void k_dinv(const float* __restrict__ adj, const float* __restrict__ mf,
                       float* __restrict__ dinv) {
  int idx = blockIdx.x * blockDim.x + threadIdx.x;
  if (idx >= TT * NNODES) return;
  int t = idx / NNODES, j = idx % NNODES;
  const float* ac = adj + (size_t)t * NNODES * NNODES + j;
  const float* m = mf + (size_t)t * NNODES;
  float s = 0.0f;
  for (int i = 0; i < NNODES; ++i) s += ac[(size_t)i * NNODES] * m[i];
  float deg = m[j] * (s + 1.0f);
  dinv[idx] = (deg > 0.0f) ? rsqrtf(deg) : 0.0f;
}

__global__ void k_f32tof16(const float* __restrict__ src, _Float16* __restrict__ dst,
                           long n) {
  long idx = (long)blockIdx.x * blockDim.x + threadIdx.x;
  if (idx < n) dst[idx] = (_Float16)src[idx];
}

// adjT16[t][j][i] = (f16)adj[t][i][j]  (raw 0/1 adjacency, exact in f16)
__global__ void k_adjT(const float* __restrict__ adj, _Float16* __restrict__ adjT) {
  long idx = (long)blockIdx.x * blockDim.x + threadIdx.x;
  if (idx >= (long)TT * NNODES * NNODES) return;
  int i = (int)(idx & (NNODES - 1));
  long r = idx >> 9;
  int j = (int)(r & (NNODES - 1));
  int t = (int)(r >> 9);
  adjT[idx] = (_Float16)adj[(size_t)t * NNODES * NNODES + (size_t)i * NNODES + j];
}

__global__ void k_w2t(const float* __restrict__ W2, _Float16* __restrict__ W2T) {
  int idx = blockIdx.x * blockDim.x + threadIdx.x;
  if (idx >= HDIM * HDIM) return;
  int h = idx / HDIM, f = idx % HDIM;
  W2T[f * HDIM + h] = (_Float16)W2[idx];
}

// hm1T[t][f][n] = (sum_d x[t,n,d]*W1[d,f]) * dinv[t,n] * mf[t,n]
__global__ void k_mm1(const float* __restrict__ x, const float* __restrict__ W1,
                      const float* __restrict__ dinv, const float* __restrict__ mf,
                      _Float16* __restrict__ hm1T) {
  long idx = (long)blockIdx.x * blockDim.x + threadIdx.x;
  if (idx >= (long)TT * HDIM * NNODES) return;
  int n = (int)(idx & (NNODES - 1));
  long r = idx >> 9;
  int f = (int)(r & (HDIM - 1));
  int t = (int)(r >> 7);
  const float* xp = x + ((size_t)t * NNODES + n) * DINP;
  float v = xp[0] * W1[f] + xp[1] * W1[HDIM + f] + xp[2] * W1[2 * HDIM + f];
  hm1T[idx] = (_Float16)(v * dinv[t * NNODES + n] * mf[t * NNODES + n]);
}

// pad-transpose: dstT[n][o][tp] = (tp<TT) ? src[n][tp][o] : 0   (tp in [0,64))
__global__ void k_padT(const _Float16* __restrict__ src, _Float16* __restrict__ dstT) {
  long idx = (long)blockIdx.x * blockDim.x + threadIdx.x;
  if (idx >= (long)NNODES * HDIM * 64) return;
  int tp = (int)(idx & 63);
  long r = idx >> 6;
  int o = (int)(r & (HDIM - 1));
  int n = (int)(r >> 7);
  dstT[idx] = (tp < TT)
                  ? src[(size_t)n * TT * HDIM + (size_t)tp * HDIM + o]
                  : (_Float16)0.0f;
}

// ---------------- WMMA graph aggregation ----------------
// acc[j,f]  = sum_i adjT[t][j][i] * hmT[t][f][i]         (WMMA)
// out[t,j,f]= dinv[t,j]*(acc + hmT[t][f][j]) + mf[t,j]*bias[f]
// A panel (16 x 512 f16, reused by 8 waves x 16 K-chunks) staged to LDS by TDM.
__global__ void k_agg(const _Float16* __restrict__ adjT, const float* __restrict__ mf,
                      const float* __restrict__ dinv, const _Float16* __restrict__ hmT,
                      const float* __restrict__ bias, _Float16* __restrict__ out,
                      int relu, int node_major) {
  int t = blockIdx.y;
  int j0 = blockIdx.x * 16;
  int tid = threadIdx.x, lane = tid & 31, wave = tid >> 5;
  int f0 = wave * 16;
  __shared__ __align__(16) _Float16 a_lds[16 * NNODES];  // [j_local][i]

  const _Float16* apanel = adjT + (size_t)t * NNODES * NNODES + (size_t)j0 * NNODES;
#if defined(__has_builtin) && __has_builtin(__builtin_amdgcn_tensor_load_to_lds) && \
    __has_builtin(__builtin_amdgcn_s_wait_tensorcnt)
  if (wave == 0) {
    unsigned long long ga = (unsigned long long)(size_t)apanel;
    unsigned lds_off = (unsigned)(size_t)(void*)a_lds;  // low 32 bits = LDS offset
    // D# group0: count=1 | lds_addr | global_addr[56:0] | type=2
    u32x4 g0;
    g0[0] = 1u;
    g0[1] = lds_off;
    g0[2] = (unsigned)(ga & 0xffffffffull);
    g0[3] = (unsigned)((ga >> 32) & 0x1ffffffull) | (2u << 30);
    // D# group1: wg_mask=0, data_size=1(2B), tensor_dim0=512, tensor_dim1=512,
    // tile_dim0=512, tile_dim1=16, tile_dim2=0, tensor_dim0_stride=512
    i32x8 g1;
    g1[0] = (int)(1u << 16);
    g1[1] = (int)((NNODES & 0xffffu) << 16);          // tensor_dim0 lo16
    g1[2] = (int)((NNODES & 0xffffu) << 16);          // td0 hi16=0 | td1 lo16
    g1[3] = (int)((unsigned)NNODES << 16);            // td1 hi16=0 | tile_dim0
    g1[4] = 16;                                       // tile_dim1=16, tile_dim2=0
    g1[5] = NNODES;                                   // tensor_dim0_stride lo32
    g1[6] = 0;
    g1[7] = 0;
    i32x4 gz = {0, 0, 0, 0};
    i32x8 gz8 = {0, 0, 0, 0, 0, 0, 0, 0};
    // 6-arg form (clang-23 / therock-10.0 headers): g0, g1, g2, g3, g4, cpol
    __builtin_amdgcn_tensor_load_to_lds(g0, g1, gz, gz, gz8, 0);
    __builtin_amdgcn_s_wait_tensorcnt(0);
  }
#else
  {
    const u32x4* src = (const u32x4*)apanel;
    u32x4* dst = (u32x4*)a_lds;
    for (int s = tid; s < 16 * NNODES / 8; s += 256) dst[s] = src[s];
  }
#endif
  __syncthreads();

  const _Float16* bbase = hmT + (size_t)t * HDIM * NNODES + (size_t)f0 * NNODES;
  v8f c = acc_zero();
  for (int kb = 0; kb < NNODES; kb += 32) {
    v16h a = frag_rowmajor_vec(a_lds + kb, NNODES, lane, 16);   // ds_load_b128 x2
    v16h b = frag_rowmajor_vec(bbase + kb, NNODES, lane, 16);   // global b128 x2
    c = wmma16(a, b, c);
  }

  int col = lane & 15, half = lane >> 4;
#pragma unroll
  for (int r = 0; r < 8; ++r) {
    int j = j0 + r + 8 * half;
    int f = f0 + col;
    float hm_self = (float)hmT[(size_t)t * HDIM * NNODES + (size_t)f * NNODES + j];
    float val = dinv[t * NNODES + j] * (c[r] + hm_self) + mf[t * NNODES + j] * bias[f];
    if (relu) val = fmaxf(val, 0.0f);
    size_t o = node_major ? ((size_t)j * TT * HDIM + (size_t)t * HDIM + f)
                          : (((size_t)t * NNODES + j) * HDIM + f);
    out[o] = (_Float16)val;
  }
}

// hm2T[t][f][n] = (sum_h g1[t,n,h]*W2[h,f]) * dinv[t,n] * mf[t,n]
__global__ void k_mm2(const _Float16* __restrict__ g1, const _Float16* __restrict__ w2t,
                      const float* __restrict__ dinv, const float* __restrict__ mf,
                      _Float16* __restrict__ hm2T) {
  int t = blockIdx.y;
  int n0 = blockIdx.x * 16;
  int tid = threadIdx.x, lane = tid & 31, wave = tid >> 5;
  int f0 = wave * 16;
  const _Float16* abase = g1 + ((size_t)t * NNODES + n0) * HDIM;
  v8f c = acc_zero();
#pragma unroll
  for (int kb = 0; kb < HDIM; kb += 32) {
    v16h a = frag_rowmajor_vec(abase + kb, HDIM, lane, 16);
    v16h b = frag_rowmajor_vec(w2t + (size_t)f0 * HDIM + kb, HDIM, lane, 16);
    c = wmma16(a, b, c);
  }
  int col = lane & 15, half = lane >> 4;
#pragma unroll
  for (int r = 0; r < 8; ++r) {
    int n = n0 + r + 8 * half;
    float s = dinv[t * NNODES + n] * mf[t * NNODES + n];
    hm2T[(size_t)t * HDIM * NNODES + (size_t)(f0 + col) * NNODES + n] =
        (_Float16)(c[r] * s);
  }
}

// ---------------- per-node linear: out[n,t,o] = (in[n,t,:] . W[n,o,:] + b[n,o]) * scale
__global__ void k_node_linear(const _Float16* __restrict__ in,
                              const _Float16* __restrict__ W, long wstride, int wrowoff,
                              const float* __restrict__ bias, int bstride, int boff,
                              _Float16* __restrict__ out16, float* __restrict__ out32,
                              float scale) {
  int n = blockIdx.x;
  int tid = threadIdx.x, lane = tid & 31, wave = tid >> 5;
  int ncol0 = wave * 16;
  const _Float16* inp = in + (size_t)n * TT * HDIM;
  const _Float16* wb = W + (size_t)n * wstride + (size_t)(wrowoff + ncol0) * HDIM;
  const float* bp = bias + (size_t)n * bstride + boff + ncol0;
  int col = lane & 15, half = lane >> 4;
  float bv = bp[col];

  v16h bf[4];
#pragma unroll
  for (int kc = 0; kc < 4; ++kc)
    bf[kc] = frag_rowmajor_vec(wb + kc * 32, HDIM, lane, 16);

  for (int mt = 0; mt < 4; ++mt) {
    int m0 = mt * 16;
    int mvalid = imin(16, TT - m0);
    v8f c = acc_zero();
#pragma unroll
    for (int kc = 0; kc < 4; ++kc) {
      v16h a = frag_rowmajor_vec(inp + (size_t)m0 * HDIM + kc * 32, HDIM, lane, mvalid);
      c = wmma16(a, bf[kc], c);
    }
#pragma unroll
    for (int r = 0; r < 8; ++r) {
      int trow = m0 + r + 8 * half;
      if (trow < TT) {
        float val = (c[r] + bv) * scale;
        size_t o = (size_t)n * TT * HDIM + (size_t)trow * HDIM + ncol0 + col;
        if (out16) out16[o] = (_Float16)val;
        if (out32) out32[o] = val;
      }
    }
  }
}

// ---------------- per-node 4-head attention over T=50 ----------------
// vhT: [n][HDIM][64] pad-transposed values (zeros at t>=TT)
__global__ void k_attn(const _Float16* __restrict__ qh, const _Float16* __restrict__ kh,
                       const _Float16* __restrict__ vhT, _Float16* __restrict__ o16) {
  int n = blockIdx.x;
  int tid = threadIdx.x, lane = tid & 31, wave = tid >> 5;
  __shared__ float S[64 * 64];
  __shared__ __align__(16) _Float16 Sh[64 * 64];
  const _Float16* qn = qh + (size_t)n * TT * HDIM;
  const _Float16* kn = kh + (size_t)n * TT * HDIM;
  const _Float16* vn = vhT + (size_t)n * HDIM * 64;
  int col = lane & 15, half = lane >> 4;
  for (int h = 0; h < NHEADS; ++h) {
    int hoff = h * HD;
    // scores S[q,k] (K = HD = 32 -> single WMMA per 16x16 tile)
    for (int tile = wave; tile < 16; tile += 8) {
      int mi = tile >> 2, ni = tile & 3;
      int m0 = mi * 16, n0 = ni * 16;
      v16h a = frag_rowmajor_vec(qn + (size_t)m0 * HDIM + hoff, HDIM, lane,
                                 imin(16, TT - m0));
      v16h b = frag_rowmajor_vec(kn + (size_t)n0 * HDIM + hoff, HDIM, lane,
                                 imin(16, TT - n0));
      v8f c = acc_zero();
      c = wmma16(a, b, c);
#pragma unroll
      for (int r = 0; r < 8; ++r) S[(m0 + r + 8 * half) * 64 + n0 + col] = c[r];
    }
    __syncthreads();
    // row softmax, f16 probs, zero-pad K to 64
    if (tid < TT) {
      float mx = -1e30f;
      for (int c2 = 0; c2 < TT; ++c2) mx = fmaxf(mx, S[tid * 64 + c2]);
      float sum = 0.0f;
      for (int c2 = 0; c2 < TT; ++c2) sum += __expf(S[tid * 64 + c2] - mx);
      float inv = 1.0f / sum;
      for (int c2 = 0; c2 < 64; ++c2) {
        float p = (c2 < TT) ? __expf(S[tid * 64 + c2] - mx) * inv : 0.0f;
        Sh[tid * 64 + c2] = (_Float16)p;
      }
    }
    __syncthreads();
    // O = P @ vh_h : M=50(4 tiles) x N=32(2 tiles), K=64 (2 chunks)
    {
      int mt = wave >> 1, ntile = wave & 1;
      int m0 = mt * 16, c0 = ntile * 16;
      v8f c = acc_zero();
#pragma unroll
      for (int kb = 0; kb < 64; kb += 32) {
        v16h a = frag_rowmajor_vec(&Sh[m0 * 64 + kb], 64, lane, imin(16, TT - m0));
        v16h b = frag_rowmajor_vec(vn + (size_t)(hoff + c0) * 64 + kb, 64, lane, 16);
        c = wmma16(a, b, c);
      }
#pragma unroll
      for (int r = 0; r < 8; ++r) {
        int trow = m0 + r + 8 * half;
        if (trow < TT)
          o16[(size_t)n * TT * HDIM + (size_t)trow * HDIM + hoff + c0 + col] =
              (_Float16)c[r];
      }
    }
    __syncthreads();
  }
}

// ---------------- launch ----------------

extern "C" void kernel_launch(void* const* d_in, const int* in_sizes, int n_in,
                              void* d_out, int out_size, void* d_ws, size_t ws_size,
                              hipStream_t stream) {
  (void)in_sizes; (void)n_in; (void)out_size; (void)ws_size;
  const float* x     = (const float*)d_in[0];
  const float* adj   = (const float*)d_in[1];
  const int*   em    = (const int*)d_in[2];
  const float* W1    = (const float*)d_in[3];
  const float* b1    = (const float*)d_in[4];
  const float* W2    = (const float*)d_in[5];
  const float* b2    = (const float*)d_in[6];
  const float* Wq    = (const float*)d_in[7];
  const float* bq    = (const float*)d_in[8];
  const float* Wk    = (const float*)d_in[9];
  const float* bk    = (const float*)d_in[10];
  const float* Wv    = (const float*)d_in[11];
  const float* bv    = (const float*)d_in[12];
  const float* W_in  = (const float*)d_in[13];
  const float* b_in  = (const float*)d_in[14];
  const float* W_out = (const float*)d_in[15];
  const float* b_out = (const float*)d_in[16];
  float* out = (float*)d_out;

  char* ws = (char*)d_ws;
  size_t off = 0;
  auto carve = [&](size_t bytes) -> void* {
    void* p = ws + off;
    off = (off + bytes + 255) & ~(size_t)255;
    return p;
  };

  const size_t NE = (size_t)TT * NNODES * HDIM;  // 3,276,800
  float*    mf    = (float*)carve((size_t)TT * NNODES * 4);
  float*    dinv  = (float*)carve((size_t)TT * NNODES * 4);
  _Float16* adjT  = (_Float16*)carve((size_t)TT * NNODES * NNODES * 2);
  _Float16* hm1T  = (_Float16*)carve(NE * 2);
  _Float16* g1    = (_Float16*)carve(NE * 2);
  _Float16* hm2T  = (_Float16*)carve(NE * 2);
  _Float16* emb   = (_Float16*)carve(NE * 2);
  _Float16* qb    = (_Float16*)carve(NE * 2);
  _Float16* kb    = (_Float16*)carve(NE * 2);
  _Float16* vb    = (_Float16*)carve(NE * 2);
  _Float16* qhb   = (_Float16*)carve(NE * 2);
  _Float16* khb   = (_Float16*)carve(NE * 2);
  _Float16* vhb   = (_Float16*)carve(NE * 2);
  _Float16* vhT   = (_Float16*)carve((size_t)NNODES * HDIM * 64 * 2);
  _Float16* ob    = (_Float16*)carve(NE * 2);
  _Float16* W2T   = (_Float16*)carve((size_t)HDIM * HDIM * 2);
  _Float16* Wqf   = (_Float16*)carve((size_t)NNODES * HDIM * HDIM * 2);
  _Float16* Wkf   = (_Float16*)carve((size_t)NNODES * HDIM * HDIM * 2);
  _Float16* Wvf   = (_Float16*)carve((size_t)NNODES * HDIM * HDIM * 2);
  _Float16* Winf  = (_Float16*)carve((size_t)NNODES * 3 * HDIM * HDIM * 2);
  _Float16* Woutf = (_Float16*)carve((size_t)NNODES * HDIM * HDIM * 2);

  const int thr = 256;
  int gTN = (TT * NNODES + thr - 1) / thr;
  k_mask<<<gTN, thr, 0, stream>>>(em, mf);
  k_dinv<<<gTN, thr, 0, stream>>>(adj, mf, dinv);

  long nadj = (long)TT * NNODES * NNODES;
  k_adjT<<<(int)((nadj + thr - 1) / thr), thr, 0, stream>>>(adj, adjT);

  long nw = (long)NNODES * HDIM * HDIM;
  k_f32tof16<<<(int)((nw + thr - 1) / thr), thr, 0, stream>>>(Wq, Wqf, nw);
  k_f32tof16<<<(int)((nw + thr - 1) / thr), thr, 0, stream>>>(Wk, Wkf, nw);
  k_f32tof16<<<(int)((nw + thr - 1) / thr), thr, 0, stream>>>(Wv, Wvf, nw);
  long nwi = (long)NNODES * 3 * HDIM * HDIM;
  k_f32tof16<<<(int)((nwi + thr - 1) / thr), thr, 0, stream>>>(W_in, Winf, nwi);
  k_f32tof16<<<(int)((nw + thr - 1) / thr), thr, 0, stream>>>(W_out, Woutf, nw);
  k_w2t<<<(HDIM * HDIM + thr - 1) / thr, thr, 0, stream>>>(W2, W2T);

  k_mm1<<<(int)((NE + thr - 1) / thr), thr, 0, stream>>>(x, W1, dinv, mf, hm1T);

  dim3 gagg(NNODES / 16, TT);
  k_agg<<<gagg, thr, 0, stream>>>(adjT, mf, dinv, hm1T, b1, g1, /*relu=*/1, /*node_major=*/0);
  k_mm2<<<gagg, thr, 0, stream>>>(g1, W2T, dinv, mf, hm2T);
  k_agg<<<gagg, thr, 0, stream>>>(adjT, mf, dinv, hm2T, b2, emb, /*relu=*/0, /*node_major=*/1);

  const float sc = 0.17677669529663689f;  // HD^-0.5
  k_node_linear<<<NNODES, thr, 0, stream>>>(emb, Wqf, HDIM * HDIM, 0, bq, HDIM, 0, qb, nullptr, 1.0f);
  k_node_linear<<<NNODES, thr, 0, stream>>>(emb, Wkf, HDIM * HDIM, 0, bk, HDIM, 0, kb, nullptr, 1.0f);
  k_node_linear<<<NNODES, thr, 0, stream>>>(emb, Wvf, HDIM * HDIM, 0, bv, HDIM, 0, vb, nullptr, 1.0f);
  k_node_linear<<<NNODES, thr, 0, stream>>>(qb, Winf, 3 * HDIM * HDIM, 0,        b_in, 3 * HDIM, 0,        qhb, nullptr, sc);
  k_node_linear<<<NNODES, thr, 0, stream>>>(kb, Winf, 3 * HDIM * HDIM, HDIM,     b_in, 3 * HDIM, HDIM,     khb, nullptr, 1.0f);
  k_node_linear<<<NNODES, thr, 0, stream>>>(vb, Winf, 3 * HDIM * HDIM, 2 * HDIM, b_in, 3 * HDIM, 2 * HDIM, vhb, nullptr, 1.0f);

  long npad = (long)NNODES * HDIM * 64;
  k_padT<<<(int)((npad + thr - 1) / thr), thr, 0, stream>>>(vhb, vhT);

  k_attn<<<NNODES, thr, 0, stream>>>(qhb, khb, vhT, ob);

  k_node_linear<<<NNODES, thr, 0, stream>>>(ob, Woutf, HDIM * HDIM, 0, b_out, HDIM, 0, nullptr, out, 1.0f);
}
